// Attention_14740327760226
// MI455X (gfx1250) — matrix operand order repaired and checked by reading, not compiled, not run on hardware
//
#include <hip/hip_runtime.h>
#include <hip/hip_bf16.h>

typedef __bf16 bf16;
typedef bf16  v16bf __attribute__((ext_vector_type(16)));
typedef bf16  v8bf  __attribute__((ext_vector_type(8)));
typedef float v8f   __attribute__((ext_vector_type(8)));
typedef unsigned int v4u __attribute__((ext_vector_type(4)));
typedef int  v8i __attribute__((ext_vector_type(8)));
typedef int  v4i __attribute__((ext_vector_type(4)));

#define CAT16(lo, hi) __builtin_shufflevector((lo), (hi), 0,1,2,3,4,5,6,7,8,9,10,11,12,13,14,15)

#if __has_builtin(__builtin_amdgcn_tensor_load_to_lds) && __has_builtin(__builtin_amdgcn_s_wait_tensorcnt)
#define HAVE_TDM 1
#else
#define HAVE_TDM 0
#endif

static constexpr int SEQ  = 4096;
static constexpr int DM   = 768;
static constexpr int NH   = 12;
static constexpr int HD   = 64;
static constexpr int DQKV = 3 * DM;   // 2304

// ---------------------------------------------------------------------------
// fp32 -> bf16 elementwise convert
// ---------------------------------------------------------------------------
__global__ __launch_bounds__(256) void to_bf16_kernel(const float* __restrict__ src,
                                                      bf16* __restrict__ dst, long n) {
  long i = (long)blockIdx.x * blockDim.x + threadIdx.x;
  if (i < n) dst[i] = (bf16)src[i];
}

// src [R][C] fp32  ->  dst [C][R] bf16   (dst writes coalesced)
__global__ __launch_bounds__(256) void transpose_to_bf16_kernel(const float* __restrict__ src,
                                                                bf16* __restrict__ dst,
                                                                int R, int C) {
  long i = (long)blockIdx.x * blockDim.x + threadIdx.x;
  long total = (long)R * C;
  if (i < total) {
    int n = (int)(i / R);   // dst row (original column)
    int k = (int)(i % R);   // dst col (original row)
    dst[i] = (bf16)src[(long)k * C + n];
  }
}

#if HAVE_TDM
// ---------------------------------------------------------------------------
// Issue one TDM 2D tile load: tile_rows x tile_k bf16 from a [*][tensor_k]
// row-major tensor into LDS with 16B padding after every 64B of data
// (LDS row stride = 40 bf16). Descriptor per CDNA5 ISA 8.3/8.4 (type=2).
// This toolchain exposes the 6-arg builtin:
//   (uint32x4 g0, int32x8 g1, int32x4 g2, int32x4 g3, int32x8 g4, i32 cpol)
// Groups 2/3 (3D+/gather) and the extra group are unused -> zero-filled.
// ---------------------------------------------------------------------------
__device__ __forceinline__ void tdm_load_tile(const bf16* gptr, const bf16* lds,
                                              int tensor_k, int tensor_rows,
                                              int tile_k, int tile_rows) {
  unsigned long long ga = (unsigned long long)(size_t)(const void*)gptr;
  unsigned lds_off = (unsigned)(size_t)(const void*)lds;   // low 32b of flat = LDS byte offset
  v4u g0;
  g0[0] = 1u;                                            // count=1 (valid user descriptor)
  g0[1] = lds_off;                                       // lds_addr
  g0[2] = (unsigned)(ga & 0xFFFFFFFFu);                  // global_addr[31:0]
  g0[3] = (unsigned)((ga >> 32) & 0x1FFFFFFu) | (2u << 30);  // global_addr[56:32] | type=2
  v8i g1;
  g1[0] = (int)((1u << 16)     // data_size = 2 bytes (bf16)
              | (1u << 20)     // pad_enable
              | (3u << 22)     // pad_interval: 16 DWORDs (=64B of data)
              | (3u << 25));   // pad_amount:   4 DWORDs (=16B pad) -> LDS stride 80B
  g1[1] = (int)(((unsigned)tensor_k & 0xFFFFu) << 16);               // tensor_dim0[15:0]
  g1[2] = (int)(((unsigned)tensor_k >> 16) |
                (((unsigned)tensor_rows & 0xFFFFu) << 16));          // dim0 hi | dim1 lo
  g1[3] = (int)(((unsigned)tensor_rows >> 16) |
                ((unsigned)tile_k << 16));                           // dim1 hi | tile_dim0
  g1[4] = (int)((unsigned)tile_rows & 0xFFFFu);                      // tile_dim1 (tile_dim2=0)
  g1[5] = tensor_k;                                                  // tensor_dim0_stride[31:0]
  g1[6] = 0;
  g1[7] = 0;
  v4i z4 = {};
  v8i z8 = {};
  __builtin_amdgcn_tensor_load_to_lds(g0, g1, z4, z4, z8, 0);
}
#endif

// ---------------------------------------------------------------------------
// TN GEMM, bf16 inputs, fp32 accumulate via v_wmma_f32_16x16x32_bf16.
// A: [M][K] row-major bf16.  Bt: [N][K] row-major bf16 (pre-transposed weight).
// Block tile 128x128, 8 waves, wave tile 64(M) x 32(N), K step 32.
// Staging: TDM (tensor_load_to_lds) double-buffered when available.
// MODE 0: Cout[row][col] = acc + bias[col]  (fp32)
// MODE 1: QKV epilogue -> Qh (scaled 1/8), Kh, Vt (all bf16)
// ---------------------------------------------------------------------------
template <int MODE>
__global__ __launch_bounds__(256) void gemm_bf16_kernel(
    const bf16* __restrict__ A, const bf16* __restrict__ Bt,
    const float* __restrict__ bias, float* __restrict__ Cout,
    bf16* __restrict__ Qh, bf16* __restrict__ Kh, bf16* __restrict__ Vt,
    int M, int N, int K) {
  constexpr int BM = 128, BN = 128, KT = 32;
  constexpr int LDA = 40;  // 32 data + 8 pad (bf16), 80B rows -> every frag 16B aligned

  const int tid  = threadIdx.x;
  const int lane = tid & 31;
  const int wv   = tid >> 5;      // 0..7
  const int hlf  = lane >> 4;     // lane half (0/1)
  const int l16  = lane & 15;
  const int wm   = wv & 1;        // 2 waves along M
  const int wn   = wv >> 1;       // 4 waves along N
  const int bm0  = blockIdx.y * BM;
  const int bn0  = blockIdx.x * BN;

  v8f acc[4][2] = {};

  auto compute_tile = [&](const bf16* Asrc, const bf16* Bsrc) {
    v16bf bfrag[2];
#pragma unroll
    for (int n = 0; n < 2; ++n) {
      const bf16* bp = Bsrc + (wn * 32 + n * 16 + l16) * LDA;
      v8bf lo = *(const v8bf*)(bp + hlf * 8);
      v8bf hi = *(const v8bf*)(bp + 16 + hlf * 8);
      bfrag[n] = CAT16(lo, hi);
    }
#pragma unroll
    for (int m = 0; m < 4; ++m) {
      const bf16* ap = Asrc + (wm * 64 + m * 16 + l16) * LDA;
      v8bf lo = *(const v8bf*)(ap + hlf * 8);
      v8bf hi = *(const v8bf*)(ap + 16 + hlf * 8);
      v16bf afrag = CAT16(lo, hi);
#pragma unroll
      for (int n = 0; n < 2; ++n)
        acc[m][n] = __builtin_amdgcn_wmma_f32_16x16x32_bf16(
            false, afrag, false, bfrag[n], (short)0, acc[m][n], false, false);
    }
  };

#if HAVE_TDM
  __shared__ __align__(16) bf16 As[2][BM * LDA];
  __shared__ __align__(16) bf16 Bs[2][BN * LDA];
  const int nk = K / KT;
  if (wv == 0) {
    tdm_load_tile(A + (long)bm0 * K, As[0], K, M, KT, BM);
    tdm_load_tile(Bt + (long)bn0 * K, Bs[0], K, N, KT, BN);
  }
  for (int kt = 0; kt < nk; ++kt) {
    const int cur = kt & 1;
    if (wv == 0) {
      if (kt + 1 < nk) {
        tdm_load_tile(A + (long)bm0 * K + (kt + 1) * KT, As[cur ^ 1], K, M, KT, BM);
        tdm_load_tile(Bt + (long)bn0 * K + (kt + 1) * KT, Bs[cur ^ 1], K, N, KT, BN);
        __builtin_amdgcn_s_wait_tensorcnt((short)2);  // current tile's 2 DMAs done (in-order)
      } else {
        __builtin_amdgcn_s_wait_tensorcnt((short)0);
      }
    }
    __syncthreads();                 // release all waves to read buffer `cur`
    compute_tile(cur ? As[1] : As[0], cur ? Bs[1] : Bs[0]);
    __syncthreads();                 // everyone done with `cur` before it is re-filled
  }
#else
  __shared__ __align__(16) bf16 As[BM * LDA];
  __shared__ __align__(16) bf16 Bs[BN * LDA];
  for (int k0 = 0; k0 < K; k0 += KT) {
    __syncthreads();
#pragma unroll
    for (int it = 0; it < 2; ++it) {
      int c = tid + it * 256;        // 512 chunks of 16B per tile
      int r = c >> 2, cc = c & 3;
      *(v8bf*)(As + r * LDA + cc * 8) =
          *(const v8bf*)(A + (long)(bm0 + r) * K + k0 + cc * 8);
    }
#pragma unroll
    for (int it = 0; it < 2; ++it) {
      int c = tid + it * 256;
      int r = c >> 2, cc = c & 3;
      *(v8bf*)(Bs + r * LDA + cc * 8) =
          *(const v8bf*)(Bt + (long)(bn0 + r) * K + k0 + cc * 8);
    }
    __syncthreads();
    compute_tile(As, Bs);
  }
#endif

  // Epilogue: C layout = VGPR r holds M = r + 8*hlf, N = lane%16
#pragma unroll
  for (int m = 0; m < 4; ++m)
#pragma unroll
    for (int n = 0; n < 2; ++n)
#pragma unroll
      for (int r = 0; r < 8; ++r) {
        int row = bm0 + wm * 64 + m * 16 + r + hlf * 8;
        int col = bn0 + wn * 32 + n * 16 + l16;
        float v = acc[m][n][r] + bias[col];
        if (MODE == 0) {
          Cout[(long)row * N + col] = v;
        } else {
          int seg = col / DM;         // 0=Q 1=K 2=V
          int cm  = col - seg * DM;
          int h = cm >> 6, d = cm & 63;
          if (seg == 0)
            Qh[((long)h * SEQ + row) * HD + d] = (bf16)(v * 0.125f);  // fold 1/sqrt(64)
          else if (seg == 1)
            Kh[((long)h * SEQ + row) * HD + d] = (bf16)v;
          else
            Vt[((long)h * HD + d) * SEQ + row] = (bf16)v;
        }
      }
}

// ---------------------------------------------------------------------------
// Flash attention: block = (q-tile of 64, head). 4 waves x 16 q-rows each.
// Qh[h][s][64] (pre-scaled), Kh[h][s][64], Vt[h][64][s] -> Abf[s][768] bf16
// ---------------------------------------------------------------------------
__global__ __launch_bounds__(128) void flash_attn_kernel(
    const bf16* __restrict__ Qh, const bf16* __restrict__ Kh,
    const bf16* __restrict__ Vt, bf16* __restrict__ Abf) {
  const int h   = blockIdx.y;
  const int qt  = blockIdx.x;
  const int tid = threadIdx.x;
  const int lane = tid & 31;
  const int wv   = tid >> 5;      // 0..3
  const int hlf  = lane >> 4;
  const int l16  = lane & 15;
  const int qrow0 = qt * 64 + wv * 16;

  __shared__ __align__(16) bf16 Pl[4][16][72];  // per-wave P tile, 144B rows (16B aligned)

  // Q A-fragments (K = head dim, 2 chunks of 32)
  const bf16* qbase = Qh + ((long)h * SEQ + qrow0 + l16) * HD;
  v16bf qfrag[2];
#pragma unroll
  for (int c = 0; c < 2; ++c) {
    v8bf lo = *(const v8bf*)(qbase + c * 32 + hlf * 8);
    v8bf hi = *(const v8bf*)(qbase + c * 32 + 16 + hlf * 8);
    qfrag[c] = CAT16(lo, hi);
  }

  float mrun[8], lrun[8];
  v8f o[4] = {};
#pragma unroll
  for (int r = 0; r < 8; ++r) { mrun[r] = -3.0e38f; lrun[r] = 0.0f; }

  for (int j = 0; j <= qt; ++j) {
    const int kb = j * 64;

    // pull next key/value tile toward the WGP while this one computes
    if (j < qt) {
      __builtin_prefetch(Kh + ((long)h * SEQ + kb + 64 + lane * 2) * HD, 0, 1);
      __builtin_prefetch(Vt + ((long)h * HD + lane * 2) * SEQ + kb + 64, 0, 1);
    }

    // ---- S = Q @ K^T (16x64 per wave) ----
    v8f s[4] = {};
#pragma unroll
    for (int c = 0; c < 2; ++c) {
#pragma unroll
      for (int t = 0; t < 4; ++t) {
        const bf16* kp = Kh + ((long)h * SEQ + kb + t * 16 + l16) * HD + c * 32;
        v8bf lo = *(const v8bf*)(kp + hlf * 8);
        v8bf hi = *(const v8bf*)(kp + 16 + hlf * 8);
        v16bf bfrag = CAT16(lo, hi);
        s[t] = __builtin_amdgcn_wmma_f32_16x16x32_bf16(
            false, qfrag[c], false, bfrag, (short)0, s[t], false, false);
      }
    }

    // ---- causal mask (only diagonal tile needs it) ----
    if (j == qt) {
#pragma unroll
      for (int t = 0; t < 4; ++t)
#pragma unroll
        for (int r = 0; r < 8; ++r) {
          int qrow = qrow0 + r + hlf * 8;
          int key  = kb + t * 16 + l16;
          if (key > qrow) s[t][r] = -1.0e10f;
        }
    }

    // ---- online softmax: row stats across 4 N-subtiles + 16 lanes ----
#pragma unroll
    for (int r = 0; r < 8; ++r) {
      float v = fmaxf(fmaxf(s[0][r], s[1][r]), fmaxf(s[2][r], s[3][r]));
#pragma unroll
      for (int off = 1; off < 16; off <<= 1) v = fmaxf(v, __shfl_xor(v, off, 16));
      float mnew = fmaxf(mrun[r], v);
      float corr = __expf(mrun[r] - mnew);
      mrun[r] = mnew;
      float rsum = 0.0f;
#pragma unroll
      for (int t = 0; t < 4; ++t) {
        float p = __expf(s[t][r] - mnew);
        s[t][r] = p;
        rsum += p;
      }
#pragma unroll
      for (int off = 1; off < 16; off <<= 1) rsum += __shfl_xor(rsum, off, 16);
      lrun[r] = lrun[r] * corr + rsum;
#pragma unroll
      for (int t = 0; t < 4; ++t) o[t][r] *= corr;
    }

    // ---- P (bf16) -> LDS to re-swizzle C-layout -> A-layout ----
#pragma unroll
    for (int t = 0; t < 4; ++t)
#pragma unroll
      for (int r = 0; r < 8; ++r)
        Pl[wv][r + hlf * 8][t * 16 + l16] = (bf16)s[t][r];
    asm volatile("s_wait_dscnt 0" ::: "memory");   // wave-local LDS RAW fence

    // ---- O += P @ V ----
#pragma unroll
    for (int c = 0; c < 2; ++c) {
      const bf16* pp = &Pl[wv][l16][0] + c * 32;
      v8bf plo = *(const v8bf*)(pp + hlf * 8);
      v8bf phi = *(const v8bf*)(pp + 16 + hlf * 8);
      v16bf afrag = CAT16(plo, phi);
#pragma unroll
      for (int t = 0; t < 4; ++t) {
        const bf16* vp = Vt + ((long)h * HD + t * 16 + l16) * SEQ + kb + c * 32;
        v8bf vlo = *(const v8bf*)(vp + hlf * 8);
        v8bf vhi = *(const v8bf*)(vp + 16 + hlf * 8);
        v16bf bfrag = CAT16(vlo, vhi);
        o[t] = __builtin_amdgcn_wmma_f32_16x16x32_bf16(
            false, afrag, false, bfrag, (short)0, o[t], false, false);
      }
    }
  }

  // ---- finalize: O / l, write heads interleaved back to [S][768] ----
#pragma unroll
  for (int t = 0; t < 4; ++t)
#pragma unroll
    for (int r = 0; r < 8; ++r) {
      float v = o[t][r] / lrun[r];
      int row = qrow0 + r + hlf * 8;
      int col = h * HD + t * 16 + l16;
      Abf[(long)row * DM + col] = (bf16)v;
    }
}

// ---------------------------------------------------------------------------
extern "C" void kernel_launch(void* const* d_in, const int* in_sizes, int n_in,
                              void* d_out, int out_size, void* d_ws, size_t ws_size,
                              hipStream_t stream) {
  (void)in_sizes; (void)n_in; (void)out_size; (void)ws_size;
  const float* tokens    = (const float*)d_in[0];
  // d_in[1] = attn_bias (causal mask) — causality computed analytically
  const float* c_attn_w  = (const float*)d_in[2];
  const float* c_attn_b  = (const float*)d_in[3];
  const float* c_proj_w  = (const float*)d_in[4];
  const float* c_proj_b  = (const float*)d_in[5];
  float* out = (float*)d_out;

  char* ws = (char*)d_ws;
  bf16* tok_bf  = (bf16*)ws;  ws += (long)SEQ * DM * 2;      // 6.29 MB
  bf16* wqkv_t  = (bf16*)ws;  ws += (long)DQKV * DM * 2;     // 3.54 MB
  bf16* wproj_t = (bf16*)ws;  ws += (long)DM * DM * 2;       // 1.18 MB
  bf16* Qh      = (bf16*)ws;  ws += (long)NH * SEQ * HD * 2; // 6.29 MB
  bf16* Kh      = (bf16*)ws;  ws += (long)NH * SEQ * HD * 2;
  bf16* Vt      = (bf16*)ws;  ws += (long)NH * SEQ * HD * 2;
  bf16* Abf     = (bf16*)ws;  ws += (long)SEQ * DM * 2;      // total ~36 MB

  // 1) convert tokens to bf16
  {
    long n = (long)SEQ * DM;
    to_bf16_kernel<<<dim3((unsigned)((n + 255) / 256)), 256, 0, stream>>>(tokens, tok_bf, n);
  }
  // 2) pre-transpose weights to [N][K] bf16
  {
    long n = (long)DM * DQKV;
    transpose_to_bf16_kernel<<<dim3((unsigned)((n + 255) / 256)), 256, 0, stream>>>(
        c_attn_w, wqkv_t, DM, DQKV);
  }
  {
    long n = (long)DM * DM;
    transpose_to_bf16_kernel<<<dim3((unsigned)((n + 255) / 256)), 256, 0, stream>>>(
        c_proj_w, wproj_t, DM, DM);
  }
  // 3) QKV projection with fused bias + head-split epilogue
  gemm_bf16_kernel<1><<<dim3(DQKV / 128, SEQ / 128), 256, 0, stream>>>(
      tok_bf, wqkv_t, c_attn_b, nullptr, Qh, Kh, Vt, SEQ, DQKV, DM);
  // 4) causal flash attention
  flash_attn_kernel<<<dim3(SEQ / 64, NH), 128, 0, stream>>>(Qh, Kh, Vt, Abf);
  // 5) output projection + bias -> fp32 d_out
  gemm_bf16_kernel<0><<<dim3(DM / 128, SEQ / 128), 256, 0, stream>>>(
      Abf, wproj_t, c_proj_b, out, nullptr, nullptr, nullptr, SEQ, DM, DM);
}
